// MultiHeadAttention_47923245089261
// MI455X (gfx1250) — matrix operand order repaired and checked by reading, not compiled, not run on hardware
//
#include <hip/hip_runtime.h>

// MI455X / gfx1250, wave32. All matmuls via v_wmma_f32_16x16x32_f16.
// Pipeline:
//   [pack x/qkv_w/out_w -> f16] -> [gemm_qkv f16xf16 -> f16]
//   -> [vsum] -> [flash_attn (64-key LDS-staged tiles, WMMA row-sums)]
//   -> [gemm_out f16xf16 -> f32]

typedef __attribute__((ext_vector_type(16))) _Float16 v16h;
typedef __attribute__((ext_vector_type(8)))  _Float16 v8h;
typedef __attribute__((ext_vector_type(8)))  float    v8f;

#define SEQ        2048
#define EMBED      1024
#define HEADS      16
#define HDIM       64
#define BATCH      2
#define QKV_STRIDE 3072   // 3*EMBED

static __device__ __forceinline__ v8f wmma_f16(v16h a, v16h b, v8f c) {
  // (neg_a, A, neg_b, B, c_mod, C, reuse_a, reuse_b)
  return __builtin_amdgcn_wmma_f32_16x16x32_f16(false, a, false, b, (short)0, c, false, false);
}

// ---------------------------------------------------------------------------
// fp32 -> fp16 pack (grid-stride)
// ---------------------------------------------------------------------------
__global__ void f32_to_f16(const float* __restrict__ in, _Float16* __restrict__ out, int n)
{
  int i = blockIdx.x * blockDim.x + threadIdx.x;
  const int stride = gridDim.x * blockDim.x;
  for (; i < n; i += stride) out[i] = (_Float16)in[i];
}

// ---------------------------------------------------------------------------
// C[M,N] = A[M,K] @ W[N,K]^T + bias[N]   (A, W pre-packed fp16)
// Block: 256 threads = 8 waves = 2(Mgrp) x 4(Ngrp). Wave tile: 32(M) x 64(N),
// 8 WMMAs per 32-k step; each B fragment reused for both M sub-tiles.
// A frag (16x32): lane row = l16, K = k0 + hi*16 + e (two b128 loads).
// B frag (32x16): lane col = l16, K = k0 + hi*16 + e (one 32-byte v16h load).
// ---------------------------------------------------------------------------
template<bool OUT_HALF>
__global__ __launch_bounds__(256)
void gemm_bias_wmma(const _Float16* __restrict__ A, const _Float16* __restrict__ W,
                    const float* __restrict__ bias, void* __restrict__ Cv,
                    int M, int N, int K)
{
  const int lane = threadIdx.x & 31;
  const int l16  = lane & 15;
  const int hi   = lane >> 4;
  const int wave = threadIdx.x >> 5;
  const int rowBase = blockIdx.y * 64 + (wave >> 2) * 32;
  const int colBase = blockIdx.x * 256 + (wave & 3) * 64;

  v8f acc[2][4] = {};
  const _Float16* arow0 = A + (size_t)(rowBase + l16) * K;
  const _Float16* arow1 = A + (size_t)(rowBase + 16 + l16) * K;

  for (int k0 = 0; k0 < K; k0 += 32) {
    __builtin_prefetch((const void*)(arow0 + k0 + 256));
    __builtin_prefetch((const void*)(arow1 + k0 + 256));

    v16h af0, af1;
    {
      const v8h a00 = *(const v8h*)(arow0 + k0 + hi * 16);
      const v8h a01 = *(const v8h*)(arow0 + k0 + 8 + hi * 16);
      const v8h a10 = *(const v8h*)(arow1 + k0 + hi * 16);
      const v8h a11 = *(const v8h*)(arow1 + k0 + 8 + hi * 16);
#pragma unroll
      for (int e = 0; e < 8; ++e) {
        af0[e] = a00[e]; af0[e + 8] = a01[e];
        af1[e] = a10[e]; af1[e + 8] = a11[e];
      }
    }

#pragma unroll
    for (int s = 0; s < 4; ++s) {
      const int n = colBase + s * 16 + l16;
      const v16h bf = *(const v16h*)(W + (size_t)n * K + k0 + hi * 16);
      acc[0][s] = wmma_f16(af0, bf, acc[0][s]);
      acc[1][s] = wmma_f16(af1, bf, acc[1][s]);
    }
  }

#pragma unroll
  for (int g = 0; g < 2; ++g) {
#pragma unroll
    for (int s = 0; s < 4; ++s) {
      const int n = colBase + s * 16 + l16;
      const float bv = bias[n];
#pragma unroll
      for (int r = 0; r < 8; ++r) {
        const int row = rowBase + g * 16 + r + hi * 8;  // C layout: M = r + hi*8
        const float v = acc[g][s][r] + bv;
        if (OUT_HALF) ((_Float16*)Cv)[(size_t)row * N + n] = (_Float16)v;
        else          ((float*)Cv)[(size_t)row * N + n] = v;
      }
    }
  }
}

// ---------------------------------------------------------------------------
// vsum[b,h,d] = sum_s V[b,s,h,d]   (for the reference's "+1e-9" softmax term)
// ---------------------------------------------------------------------------
__global__ void vsum_kernel(const _Float16* __restrict__ qkv, float* __restrict__ vsum)
{
  const int bh = blockIdx.x;            // 0..31
  const int b = bh >> 4, h = bh & 15;
  const int d = threadIdx.x;            // 0..63
  const _Float16* vb = qkv + (size_t)b * SEQ * QKV_STRIDE + 2 * EMBED + h * HDIM + d;
  float s = 0.f;
  for (int t = 0; t < SEQ; ++t) s += (float)vb[(size_t)t * QKV_STRIDE];
  vsum[bh * HDIM + d] = s;
}

// ---------------------------------------------------------------------------
// Flash attention, 64-key blocks. Block = 128 threads = 4 waves; each wave
// owns 16 q-rows. All waves share (b,h) and a uniform kb loop, so per
// iteration the block stages:
//   sK [64][64] key-major   (score B-frags: contiguous b128 LDS reads)
//   sVt[64][64] d-major     (PV B-frags: contiguous b128 LDS reads)
// Scores: 8 WMMAs. Online softmax: row max via 4 ds_bpermute steps; row sums
// via 2 WMMAs against an all-ones B (exactly consistent with the f16 P used
// in P@V). Causal masking only in the final (diagonal) iteration. PV: 8 WMMAs.
// ---------------------------------------------------------------------------
__global__ __launch_bounds__(128)
void flash_attn(const _Float16* __restrict__ qkv, const float* __restrict__ vsum,
                _Float16* __restrict__ attn)
{
  __shared__ _Float16 sK [64][64];       // 8 KB: keys x d
  __shared__ _Float16 sVt[64][64];       // 8 KB: d x keys (transposed)
  __shared__ _Float16 plds[4][16][64];   // 8 KB: per-wave P tile (16 q x 64 k)

  const int tid  = threadIdx.x;
  const int lane = tid & 31;
  const int l16  = lane & 15;
  const int hi   = lane >> 4;
  const int wave = tid >> 5;
  const int bh = blockIdx.x;             // 0..31
  const int b = bh >> 4, h = bh & 15;
  const int blockQ = blockIdx.y * 64;
  const int qBase  = blockQ + wave * 16;

  const _Float16* base = qkv + (size_t)b * SEQ * QKV_STRIDE;
  const _Float16* Qp = base + h * HDIM;              // + q*3072 + d
  const _Float16* Kp = base + EMBED + h * HDIM;      // + k*3072 + d
  const _Float16* Vp = base + 2 * EMBED + h * HDIM;  // + k*3072 + d

  // Q fragments for d-chunks 0..31 / 32..63, prescaled by 1/sqrt(64) once
  v16h qf[2];
  {
    const _Float16* qrow = Qp + (size_t)(qBase + l16) * QKV_STRIDE;
#pragma unroll
    for (int c = 0; c < 2; ++c) {
      const v8h q0 = *(const v8h*)(qrow + c * 32 + hi * 16);
      const v8h q1 = *(const v8h*)(qrow + c * 32 + 8 + hi * 16);
#pragma unroll
      for (int e = 0; e < 8; ++e) {
        qf[c][e]     = (_Float16)((float)q0[e] * 0.125f);
        qf[c][e + 8] = (_Float16)((float)q1[e] * 0.125f);
      }
    }
  }

  v16h onesB;
#pragma unroll
  for (int e = 0; e < 16; ++e) onesB[e] = (_Float16)1.0f;

  v8f acc[4] = {};
  float m_i[8], l_i[8];
#pragma unroll
  for (int r = 0; r < 8; ++r) { m_i[r] = -3.0e38f; l_i[r] = 0.f; }

  // cooperative-fill: 128 threads x 2 v16h covers one 64x64 tile
  const int fKey = tid >> 1;             // 0..63
  const int fD   = (tid & 1) * 32;       // 0, 32

  const int nkb = blockQ / 64 + 1;       // uniform across the block
  for (int kb = 0; kb < nkb; ++kb) {
    const int kstart = kb * 64;
    const bool diag = (kb == nkb - 1);   // uniform branch

    // ---- stage K and V^T tiles in LDS (shared by all 4 waves)
    __syncthreads();                     // previous iteration's LDS reads done
    {
      const _Float16* krow = Kp + (size_t)(kstart + fKey) * QKV_STRIDE + fD;
      *(v16h*)&sK[fKey][fD]      = *(const v16h*)(krow);
      *(v16h*)&sK[fKey][fD + 16] = *(const v16h*)(krow + 16);
      const _Float16* vrow = Vp + (size_t)(kstart + fKey) * QKV_STRIDE + fD;
      const v16h v0 = *(const v16h*)(vrow);
      const v16h v1 = *(const v16h*)(vrow + 16);
#pragma unroll
      for (int e = 0; e < 16; ++e) {
        sVt[fD + e][fKey]      = v0[e];
        sVt[fD + 16 + e][fKey] = v1[e];
      }
    }
    __syncthreads();

    // ---- scores: s[j] = Q @ K[kstart+16j .. +15]^T  (Q prescaled)
    v8f sc4[4] = {};
#pragma unroll
    for (int c = 0; c < 2; ++c) {
#pragma unroll
      for (int j = 0; j < 4; ++j) {
        const v16h kf = *(const v16h*)&sK[j * 16 + l16][c * 32 + hi * 16];
        sc4[j] = wmma_f16(qf[c], kf, sc4[j]);
      }
    }

    // ---- causal mask (diagonal iteration only) -> -1e9
    if (diag) {
#pragma unroll
      for (int j = 0; j < 4; ++j) {
        const int kg = kstart + j * 16 + l16;
#pragma unroll
        for (int r = 0; r < 8; ++r) {
          if (kg > qBase + r + hi * 8) sc4[j][r] = -1.0e9f;
        }
      }
    }

    // ---- online softmax: row max (4 shuffle steps), exp, rescale
    float p[4][8];
    float scale[8];
#pragma unroll
    for (int r = 0; r < 8; ++r) {
      float rm = fmaxf(fmaxf(sc4[0][r], sc4[1][r]), fmaxf(sc4[2][r], sc4[3][r]));
      rm = fmaxf(rm, __shfl_xor(rm, 1, 32));
      rm = fmaxf(rm, __shfl_xor(rm, 2, 32));
      rm = fmaxf(rm, __shfl_xor(rm, 4, 32));
      rm = fmaxf(rm, __shfl_xor(rm, 8, 32));
      const float mn = fmaxf(m_i[r], rm);
      scale[r] = __expf(m_i[r] - mn);
      m_i[r] = mn;
#pragma unroll
      for (int j = 0; j < 4; ++j) p[j][r] = __expf(sc4[j][r] - mn);
#pragma unroll
      for (int s = 0; s < 4; ++s) acc[s][r] *= scale[r];
    }

    // ---- P: C-layout -> per-wave LDS tile -> A-fragment layout
#pragma unroll
    for (int j = 0; j < 4; ++j) {
#pragma unroll
      for (int r = 0; r < 8; ++r) {
        plds[wave][r + hi * 8][j * 16 + l16] = (_Float16)p[j][r];
      }
    }
    __syncthreads();                     // orders P write -> P read (and tiles)

    v16h pf[2];
#pragma unroll
    for (int c = 0; c < 2; ++c) {
      const v8h f0 = *(const v8h*)&plds[wave][l16][c * 32 + hi * 16];
      const v8h f1 = *(const v8h*)&plds[wave][l16][c * 32 + 8 + hi * 16];
#pragma unroll
      for (int e = 0; e < 8; ++e) { pf[c][e] = f0[e]; pf[c][e + 8] = f1[e]; }
    }

    // ---- row sums via WMMA against all-ones B (consistent with f16 P)
    v8f rsum = {};
    rsum = wmma_f16(pf[0], onesB, rsum);
    rsum = wmma_f16(pf[1], onesB, rsum);
#pragma unroll
    for (int r = 0; r < 8; ++r) l_i[r] = l_i[r] * scale[r] + rsum[r];

    // ---- acc += P(16x64) @ V(64x64); V frags contiguous from transposed tile
#pragma unroll
    for (int s = 0; s < 4; ++s) {
      const int d = s * 16 + l16;
      acc[s] = wmma_f16(pf[0], *(const v16h*)&sVt[d][hi * 16],      acc[s]);
      acc[s] = wmma_f16(pf[1], *(const v16h*)&sVt[d][32 + hi * 16], acc[s]);
    }
  }

  // ---- epilogue: normalize + reference's (softmax + 1e-9) @ V correction
#pragma unroll
  for (int s = 0; s < 4; ++s) {
    const int d = s * 16 + l16;
    const float corr = 1.0e-9f * vsum[bh * HDIM + d];
#pragma unroll
    for (int r = 0; r < 8; ++r) {
      const int qg = qBase + r + hi * 8;
      const float o = acc[s][r] / l_i[r] + corr;
      attn[((size_t)b * SEQ + qg) * EMBED + h * HDIM + d] = (_Float16)o;
    }
  }
}

// ---------------------------------------------------------------------------
extern "C" void kernel_launch(void* const* d_in, const int* in_sizes, int n_in,
                              void* d_out, int out_size, void* d_ws, size_t ws_size,
                              hipStream_t stream)
{
  (void)in_sizes; (void)n_in; (void)out_size; (void)ws_size;
  const float* x     = (const float*)d_in[0];
  // d_in[1] = causal mask (handled analytically)
  const float* qkv_w = (const float*)d_in[2];
  const float* qkv_b = (const float*)d_in[3];
  const float* out_w = (const float*)d_in[4];
  const float* out_b = (const float*)d_in[5];

  char* ws = (char*)d_ws;
  const size_t MB = 1024 * 1024;
  _Float16* qkv    = (_Float16*)(ws);              // 24 MB
  _Float16* attn   = (_Float16*)(ws + 24 * MB);    //  8 MB
  float*    vsum   = (float*)   (ws + 32 * MB);    //  8 KB
  _Float16* xh     = (_Float16*)(ws + 33 * MB);    //  8 MB (4096x1024)
  _Float16* wqkvh  = (_Float16*)(ws + 41 * MB);    //  6 MB (3072x1024)
  _Float16* woh    = (_Float16*)(ws + 47 * MB);    //  2 MB (1024x1024)

  const int M = BATCH * SEQ;  // 4096

  // 0) pack fp32 operands to fp16 once
  f32_to_f16<<<dim3(2048), dim3(256), 0, stream>>>(x,     xh,    M * EMBED);
  f32_to_f16<<<dim3(2048), dim3(256), 0, stream>>>(qkv_w, wqkvh, QKV_STRIDE * EMBED);
  f32_to_f16<<<dim3(1024), dim3(256), 0, stream>>>(out_w, woh,   EMBED * EMBED);

  // 1) qkv = x @ qkv_w^T + qkv_b  (f16 x f16 -> f16)
  gemm_bias_wmma<true><<<dim3(QKV_STRIDE / 256, M / 64), dim3(256), 0, stream>>>(
      xh, wqkvh, qkv_b, qkv, M, QKV_STRIDE, EMBED);

  // 2) per-head V column sums (for the +1e-9 term)
  vsum_kernel<<<dim3(BATCH * HEADS), dim3(HDIM), 0, stream>>>(qkv, vsum);

  // 3) causal flash attention -> attn (f16, [B,S,E] with heads interleaved)
  flash_attn<<<dim3(BATCH * HEADS, SEQ / 64), dim3(128), 0, stream>>>(qkv, vsum, attn);

  // 4) out = attn @ out_w^T + out_b  (f16 x f16 -> f32)
  gemm_bias_wmma<false><<<dim3(EMBED / 256, M / 64), dim3(256), 0, stream>>>(
      attn, woh, out_b, (float*)d_out, M, EMBED, EMBED);
}
